// ECGMambaClassifier_39238821216759
// MI455X (gfx1250) — compile-verified, matrix operand order checked by reading, not run on hardware
//
#include <hip/hip_runtime.h>
#include <hip/hip_bf16.h>
#include <math.h>

// ---------------------------------------------------------------------------
// Problem constants (from the reference)
// ---------------------------------------------------------------------------
#define BB      4
#define LL      2048
#define DM      128      // D_MODEL
#define DI      256      // D_INNER
#define DS      16       // D_STATE
#define DC      4        // D_CONV
#define NL      6        // N_LAYERS
#define NC      5        // NUM_CLASSES
#define ROWS    (BB*LL)  // 8192
#define LN_EPS  1e-5f
#define NCHUNK  32       // scan chunks per sequence
#define CLEN    (LL/NCHUNK) // 64 steps per chunk

typedef __attribute__((ext_vector_type(16))) _Float16 v16h;
typedef __attribute__((ext_vector_type(8)))  _Float16 v8h;
typedef __attribute__((ext_vector_type(8)))  float    v8f;
typedef __attribute__((ext_vector_type(4)))  int      v4i;

// CDNA5 async LDS DMA (ASYNCcnt-tracked) if the toolchain exposes it.
#if __has_builtin(__builtin_amdgcn_global_load_async_to_lds_b128) && \
    __has_builtin(__builtin_amdgcn_s_wait_asynccnt)
#define HAVE_ASYNC_LDS 1
#else
#define HAVE_ASYNC_LDS 0
#endif

__device__ __forceinline__ void stage_b128(const _Float16* g, _Float16* l) {
#if HAVE_ASYNC_LDS
  __builtin_amdgcn_global_load_async_to_lds_b128(
      (__attribute__((address_space(1))) v4i*)g,
      (__attribute__((address_space(3))) v4i*)l, 0, 0);
#else
  *(v8h*)l = *(const v8h*)g;
#endif
}
__device__ __forceinline__ void wait_async_2() {
#if HAVE_ASYNC_LDS
  __builtin_amdgcn_s_wait_asynccnt(2);
#endif
}
__device__ __forceinline__ void wait_async_0() {
#if HAVE_ASYNC_LDS
  __builtin_amdgcn_s_wait_asynccnt(0);
#endif
}

// ---------------------------------------------------------------------------
// Weight conversion: (NL,K,N) f32 -> transposed (NL,N,K) f16
// ---------------------------------------------------------------------------
__global__ void transpose_to_f16_kernel(const float* __restrict__ src,
                                        _Float16* __restrict__ dst,
                                        int Kdim, int Ndim, int total) {
  int idx = blockIdx.x * 256 + threadIdx.x;
  if (idx >= total) return;
  int k   = idx % Kdim;
  int n   = (idx / Kdim) % Ndim;
  int lay = idx / (Kdim * Ndim);
  dst[idx] = (_Float16)src[(size_t)lay * Kdim * Ndim + (size_t)k * Ndim + n];
}

// ---------------------------------------------------------------------------
// Embedding: h = x*inp_w + inp_b + positional encoding  (+ f16 shadow)
// ---------------------------------------------------------------------------
__global__ void embed_kernel(const float* __restrict__ x,
                             const float* __restrict__ inp_w,
                             const float* __restrict__ inp_b,
                             float* __restrict__ h,
                             _Float16* __restrict__ h16) {
  int idx = blockIdx.x * 256 + threadIdx.x;            // over ROWS*DM
  if (idx >= ROWS * DM) return;
  int d   = idx & (DM - 1);
  int row = idx >> 7;                                  // b*LL + l
  int l   = row & (LL - 1);
  float v = x[row] * inp_w[d] + inp_b[d];
  int pair = d >> 1;
  float dv = __expf(-(float)(2 * pair) * (logf(10000.0f) / (float)DM));
  float ang = (float)l * dv;
  v += (d & 1) ? cosf(ang) : sinf(ang);
  h[idx]   = v;
  h16[idx] = (_Float16)v;
}

// ---------------------------------------------------------------------------
// WMMA GEMM: C[M,N] = A16[M,LDA](f16) * BT[N,K](f16 transposed), epilogue:
//   MODE 0: none   MODE 1: +bias   MODE 2: softplus(x + bias)
//   STORE16: also emit f16 copy of C (first LDC16 columns) for the next GEMM.
// Block = 128 threads (4 waves). Block tile = 64(M) x 32(N). K chunk = 32.
// A chunks double-buffered in LDS via async LDS DMA (ASYNCcnt) when available;
// B fragments are contiguous K-runs of transposed weights (global b128, L2 hot).
// ---------------------------------------------------------------------------
template <int K, int LDA, int MODE, bool STORE16>
__global__ void __launch_bounds__(128)
gemm_wmma_f16(const _Float16* __restrict__ A16,
              const _Float16* __restrict__ BT,
              float* __restrict__ C, int ldc,
              const float* __restrict__ bias,
              _Float16* __restrict__ C16, int ldc16) {
  __shared__ _Float16 As[2][64][40];   // 80B pitch rows (16B aligned), ping-pong

  const int tid   = threadIdx.x;
  const int lane  = tid & 31;
  const int wave  = tid >> 5;
  const int tileM = blockIdx.y * 64;
  const int tileN = blockIdx.x * 32;

  const int half = lane >> 4;
  const int mrow = lane & 15;       // A row within wave tile / B column
  const int k0   = half * 8;        // K sub-offset per documented layout

  const _Float16* bt0 = BT + (size_t)(tileN + mrow) * K;
  const _Float16* bt1 = BT + (size_t)(tileN + 16 + mrow) * K;

  v8f acc0 = {};
  v8f acc1 = {};

  // Prologue: stage chunk 0 into buffer 0 (2 x b128 per thread)
#pragma unroll
  for (int i2 = tid; i2 < 256; i2 += 128) {
    int r = i2 >> 2, s = i2 & 3;
    stage_b128(A16 + (size_t)(tileM + r) * LDA + s * 8, &As[0][r][s * 8]);
  }

  const int NIT = K / 32;
#pragma unroll
  for (int it = 0; it < NIT; ++it) {
    const int kc = it * 32;
    const int pb = it & 1;
    if (it + 1 < NIT) {
      // Overlap: DMA next chunk into the other buffer while we compute.
#pragma unroll
      for (int i2 = tid; i2 < 256; i2 += 128) {
        int r = i2 >> 2, s = i2 & 3;
        stage_b128(A16 + (size_t)(tileM + r) * LDA + (kc + 32) + s * 8,
                   &As[pb ^ 1][r][s * 8]);
      }
      wait_async_2();   // our chunk-it ops done; chunk it+1 still in flight
    } else {
      wait_async_0();
    }
    __syncthreads();    // all waves' chunk-it data landed

    // A fragment: two contiguous 8-half runs -> two ds_load_b128
    v8h alo = *(const v8h*)&As[pb][wave * 16 + mrow][k0];
    v8h ahi = *(const v8h*)&As[pb][wave * 16 + mrow][16 + k0];
    v16h afrag = __builtin_shufflevector(alo, ahi,
        0,1,2,3,4,5,6,7,8,9,10,11,12,13,14,15);

    // B fragments straight from transposed weights in global (b128)
    v8h b0lo = *(const v8h*)(bt0 + kc + k0);
    v8h b0hi = *(const v8h*)(bt0 + kc + 16 + k0);
    v8h b1lo = *(const v8h*)(bt1 + kc + k0);
    v8h b1hi = *(const v8h*)(bt1 + kc + 16 + k0);
    v16h bfrag0 = __builtin_shufflevector(b0lo, b0hi,
        0,1,2,3,4,5,6,7,8,9,10,11,12,13,14,15);
    v16h bfrag1 = __builtin_shufflevector(b1lo, b1hi,
        0,1,2,3,4,5,6,7,8,9,10,11,12,13,14,15);

    acc0 = __builtin_amdgcn_wmma_f32_16x16x32_f16(
        false, afrag, false, bfrag0, (short)0, acc0, false, false);
    acc1 = __builtin_amdgcn_wmma_f32_16x16x32_f16(
        false, afrag, false, bfrag1, (short)0, acc1, false, false);
    __syncthreads();    // buffer pb is free for re-DMA at it+2
  }

  // Store per C/D layout: acc[r] -> M = r + 8*(lane>=16), N = lane&15
  const int col    = lane & 15;
  const int rowOff = (lane >> 4) * 8;
#pragma unroll
  for (int r = 0; r < 8; ++r) {
    int row  = tileM + wave * 16 + rowOff + r;
    float v0 = acc0[r];
    float v1 = acc1[r];
    if constexpr (MODE >= 1) {
      v0 += bias[tileN + col];
      v1 += bias[tileN + 16 + col];
    }
    if constexpr (MODE == 2) {
      v0 = (v0 > 20.0f) ? v0 : log1pf(__expf(v0));
      v1 = (v1 > 20.0f) ? v1 : log1pf(__expf(v1));
    }
    C[(size_t)row * ldc + (tileN + col)]      = v0;
    C[(size_t)row * ldc + (tileN + 16 + col)] = v1;
    if constexpr (STORE16) {
      if (tileN + col < ldc16)
        C16[(size_t)row * ldc16 + (tileN + col)] = (_Float16)v0;
      if (tileN + 16 + col < ldc16)
        C16[(size_t)row * ldc16 + (tileN + 16 + col)] = (_Float16)v1;
    }
  }
}

// ---------------------------------------------------------------------------
// Depthwise causal conv (width 4) + SiLU  (+ f16 shadow)
// ---------------------------------------------------------------------------
__global__ void conv_silu_kernel(const float* __restrict__ xz,
                                 const float* __restrict__ cw,   // (DI,4)
                                 const float* __restrict__ cb,   // (DI,)
                                 float* __restrict__ xc,
                                 _Float16* __restrict__ xc16) {
  int idx = blockIdx.x * 256 + threadIdx.x;          // over ROWS*DI
  if (idx >= ROWS * DI) return;
  int d   = idx & (DI - 1);
  int row = idx >> 8;
  int b   = row >> 11;
  int l   = row & (LL - 1);
  float acc = cb[d];
#pragma unroll
  for (int k = 0; k < DC; ++k) {
    int ll = l - (DC - 1) + k;
    if (ll >= 0)
      acc += cw[d * DC + k] * xz[((size_t)(b * LL + ll)) * (2 * DI) + d];
  }
  acc = acc / (1.0f + __expf(-acc));   // SiLU
  xc[idx]   = acc;
  xc16[idx] = (_Float16)acc;
}

// ---------------------------------------------------------------------------
// Chunked selective scan (3 passes); recurrence is diagonal per (b,d,n).
// ---------------------------------------------------------------------------
__global__ void __launch_bounds__(256)
scan_pass1(const float* __restrict__ dt, const float* __restrict__ xc,
           const float* __restrict__ proj, const float* __restrict__ A_log,
           float* __restrict__ wP, float* __restrict__ wQ) {
  const int c = blockIdx.x;   // chunk
  const int b = blockIdx.y;   // batch
  const int d = threadIdx.x;
  float Arow[DS], P[DS], Q[DS];
#pragma unroll
  for (int n = 0; n < DS; ++n) {
    Arow[n] = -__expf(A_log[d * DS + n]);
    P[n] = 1.0f; Q[n] = 0.0f;
  }
  const size_t row0 = (size_t)b * LL + (size_t)c * CLEN;
  for (int l = 0; l < CLEN; ++l) {
    const size_t row = row0 + l;
    const float delta = dt[row * DI + d];
    const float xv    = xc[row * DI + d];
    const float* pr   = proj + row * (DI + 2 * DS);
#pragma unroll
    for (int n = 0; n < DS; ++n) {
      const float a   = __expf(delta * Arow[n]);
      const float dBu = delta * pr[DI + n] * xv;
      Q[n] = Q[n] * a + dBu;
      P[n] *= a;
    }
  }
  const size_t o = (((size_t)b * NCHUNK + c) * DI + d) * DS;
#pragma unroll
  for (int n = 0; n < DS; ++n) { wP[o + n] = P[n]; wQ[o + n] = Q[n]; }
}

__global__ void __launch_bounds__(256)
scan_pass2(const float* __restrict__ wP, const float* __restrict__ wQ,
           float* __restrict__ wS0) {
  const int b = blockIdx.x;
  const int d = threadIdx.x;
#pragma unroll
  for (int n = 0; n < DS; ++n) {
    float s = 0.0f;
    for (int c = 0; c < NCHUNK; ++c) {
      const size_t o = (((size_t)b * NCHUNK + c) * DI + d) * DS + n;
      wS0[o] = s;
      s = wP[o] * s + wQ[o];
    }
  }
}

__global__ void __launch_bounds__(256)
scan_pass3(const float* __restrict__ dt, const float* __restrict__ xc,
           const float* __restrict__ proj, const float* __restrict__ A_log,
           const float* __restrict__ Dp, const float* __restrict__ wS0,
           float* __restrict__ y) {
  const int c = blockIdx.x;
  const int b = blockIdx.y;
  const int d = threadIdx.x;
  float Arow[DS], state[DS];
  const size_t o = (((size_t)b * NCHUNK + c) * DI + d) * DS;
#pragma unroll
  for (int n = 0; n < DS; ++n) {
    Arow[n]  = -__expf(A_log[d * DS + n]);
    state[n] = wS0[o + n];
  }
  const float Dd = Dp[d];
  const size_t row0 = (size_t)b * LL + (size_t)c * CLEN;
  for (int l = 0; l < CLEN; ++l) {
    const size_t row = row0 + l;
    const float delta = dt[row * DI + d];
    const float xv    = xc[row * DI + d];
    const float* pr   = proj + row * (DI + 2 * DS);
    float acc = 0.0f;
#pragma unroll
    for (int n = 0; n < DS; ++n) {
      const float a   = __expf(delta * Arow[n]);
      const float dBu = delta * pr[DI + n] * xv;
      state[n] = state[n] * a + dBu;
      acc += state[n] * pr[DI + DS + n];
    }
    y[row * DI + d] = acc + xv * Dd;
  }
}

// ---------------------------------------------------------------------------
// Gate: yg16 = f16( y * silu(z) ),  z = xz[..., 256:512]
// ---------------------------------------------------------------------------
__global__ void gate_kernel(const float* __restrict__ y,
                            const float* __restrict__ xz,
                            _Float16* __restrict__ yg16) {
  int idx = blockIdx.x * 256 + threadIdx.x;          // over ROWS*DI
  if (idx >= ROWS * DI) return;
  int d   = idx & (DI - 1);
  int row = idx >> 8;
  float z = xz[(size_t)row * (2 * DI) + DI + d];
  float s = z / (1.0f + __expf(-z));
  yg16[idx] = (_Float16)(y[idx] * s);
}

// ---------------------------------------------------------------------------
// Residual add + LayerNorm over DM=128, writes h and its f16 shadow
// ---------------------------------------------------------------------------
__global__ void __launch_bounds__(128)
add_ln_kernel(const float* __restrict__ src, float* __restrict__ h,
              _Float16* __restrict__ h16,
              const float* __restrict__ w, const float* __restrict__ bvec) {
  __shared__ float red[DM];
  const int row = blockIdx.x;
  const int t   = threadIdx.x;
  float v = src[(size_t)row * DM + t] + h[(size_t)row * DM + t];
  red[t] = v;
  __syncthreads();
  for (int s = DM / 2; s > 0; s >>= 1) {
    if (t < s) red[t] += red[t + s];
    __syncthreads();
  }
  const float mean = red[0] * (1.0f / DM);
  __syncthreads();
  const float c = v - mean;
  red[t] = c * c;
  __syncthreads();
  for (int s = DM / 2; s > 0; s >>= 1) {
    if (t < s) red[t] += red[t + s];
    __syncthreads();
  }
  const float var = red[0] * (1.0f / DM);
  const float o = c * rsqrtf(var + LN_EPS) * w[t] + bvec[t];
  h[(size_t)row * DM + t]   = o;
  h16[(size_t)row * DM + t] = (_Float16)o;
}

// ---------------------------------------------------------------------------
// Mean over L
// ---------------------------------------------------------------------------
__global__ void __launch_bounds__(128)
mean_kernel(const float* __restrict__ h, float* __restrict__ hm) {
  const int b = blockIdx.x;
  const int d = threadIdx.x;
  float s = 0.0f;
  for (int l = 0; l < LL; ++l)
    s += h[((size_t)b * LL + l) * DM + d];
  hm[b * DM + d] = s * (1.0f / (float)LL);
}

// ---------------------------------------------------------------------------
// Head: LN(128) -> relu(@w1 + b1) -> @w2 + b2.  One block per batch element.
// ---------------------------------------------------------------------------
__global__ void __launch_bounds__(128)
head_kernel(const float* __restrict__ hm,
            const float* __restrict__ lw, const float* __restrict__ lb,
            const float* __restrict__ w1, const float* __restrict__ b1,
            const float* __restrict__ w2, const float* __restrict__ b2,
            float* __restrict__ out) {
  __shared__ float red[DM];
  __shared__ float hn[DM];
  __shared__ float hid[DM / 2];
  const int b = blockIdx.x;
  const int t = threadIdx.x;
  float v = hm[b * DM + t];
  red[t] = v;
  __syncthreads();
  for (int s = DM / 2; s > 0; s >>= 1) {
    if (t < s) red[t] += red[t + s];
    __syncthreads();
  }
  const float mean = red[0] * (1.0f / DM);
  __syncthreads();
  const float c = v - mean;
  red[t] = c * c;
  __syncthreads();
  for (int s = DM / 2; s > 0; s >>= 1) {
    if (t < s) red[t] += red[t + s];
    __syncthreads();
  }
  const float var = red[0] * (1.0f / DM);
  hn[t] = c * rsqrtf(var + LN_EPS) * lw[t] + lb[t];
  __syncthreads();
  if (t < DM / 2) {
    float a = b1[t];
    for (int d = 0; d < DM; ++d) a += hn[d] * w1[d * (DM / 2) + t];
    hid[t] = fmaxf(a, 0.0f);
  }
  __syncthreads();
  if (t < NC) {
    float a = b2[t];
    for (int j = 0; j < DM / 2; ++j) a += hid[j] * w2[j * NC + t];
    out[b * NC + t] = a;
  }
}

// ---------------------------------------------------------------------------
// Host orchestration
// ---------------------------------------------------------------------------
static inline char* carve(char*& p, size_t bytes) {
  char* r = p;
  p += (bytes + 255) & ~(size_t)255;
  return r;
}

extern "C" void kernel_launch(void* const* d_in, const int* in_sizes, int n_in,
                              void* d_out, int out_size, void* d_ws, size_t ws_size,
                              hipStream_t stream) {
  (void)in_sizes; (void)n_in; (void)out_size; (void)ws_size;

  const float* x        = (const float*)d_in[0];
  const float* inp_w    = (const float*)d_in[1];
  const float* inp_b    = (const float*)d_in[2];
  const float* in_proj  = (const float*)d_in[3];   // (6,128,512)
  const float* conv_w   = (const float*)d_in[4];   // (6,256,1,4)
  const float* conv_b   = (const float*)d_in[5];   // (6,256)
  const float* xproj_w  = (const float*)d_in[6];   // (6,256,288)
  const float* dt_w     = (const float*)d_in[7];   // (6,256,256)
  const float* dt_b     = (const float*)d_in[8];   // (6,256)
  const float* A_log    = (const float*)d_in[9];   // (6,256,16)
  const float* Dp       = (const float*)d_in[10];  // (6,256)
  const float* out_w    = (const float*)d_in[11];  // (6,256,128)
  const float* ln_w     = (const float*)d_in[12];  // (6,128)
  const float* ln_b     = (const float*)d_in[13];  // (6,128)
  const float* cls_ln_w = (const float*)d_in[14];
  const float* cls_ln_b = (const float*)d_in[15];
  const float* cls_w1   = (const float*)d_in[16];  // (128,64)
  const float* cls_b1   = (const float*)d_in[17];
  const float* cls_w2   = (const float*)d_in[18];  // (64,5)
  const float* cls_b2   = (const float*)d_in[19];
  float* out = (float*)d_out;                      // (4,5)

  // Workspace carving
  char* p = (char*)d_ws;
  float*    wh     = (float*)   carve(p, (size_t)ROWS * DM * 4);
  float*    wxz    = (float*)   carve(p, (size_t)ROWS * 2 * DI * 4);
  float*    wxc    = (float*)   carve(p, (size_t)ROWS * DI * 4);
  float*    wproj  = (float*)   carve(p, (size_t)ROWS * (DI + 2 * DS) * 4);
  float*    wdt    = (float*)   carve(p, (size_t)ROWS * DI * 4);
  float*    wy     = (float*)   carve(p, (size_t)ROWS * DI * 4);
  float*    wout   = (float*)   carve(p, (size_t)ROWS * DM * 4);
  float*    whm    = (float*)   carve(p, (size_t)BB * DM * 4);
  float*    wP     = (float*)   carve(p, (size_t)BB * NCHUNK * DI * DS * 4);
  float*    wQ     = (float*)   carve(p, (size_t)BB * NCHUNK * DI * DS * 4);
  float*    wS0    = (float*)   carve(p, (size_t)BB * NCHUNK * DI * DS * 4);
  _Float16* wh16   = (_Float16*)carve(p, (size_t)ROWS * DM * 2);
  _Float16* wxc16  = (_Float16*)carve(p, (size_t)ROWS * DI * 2);
  _Float16* wpj16  = (_Float16*)carve(p, (size_t)ROWS * DI * 2);  // proj[:, :256]
  _Float16* wyg16  = (_Float16*)carve(p, (size_t)ROWS * DI * 2);
  _Float16* t_inproj = (_Float16*)carve(p, (size_t)NL * DM * 2 * DI * 2);
  _Float16* t_xproj  = (_Float16*)carve(p, (size_t)NL * DI * (DI + 2 * DS) * 2);
  _Float16* t_dt     = (_Float16*)carve(p, (size_t)NL * DI * DI * 2);
  _Float16* t_out    = (_Float16*)carve(p, (size_t)NL * DI * DM * 2);

  // Transposed f16 weights (every launch; deterministic)
  {
    int n;
    n = NL * DM * (2 * DI);
    transpose_to_f16_kernel<<<(n + 255) / 256, 256, 0, stream>>>(in_proj, t_inproj, DM, 2 * DI, n);
    n = NL * DI * (DI + 2 * DS);
    transpose_to_f16_kernel<<<(n + 255) / 256, 256, 0, stream>>>(xproj_w, t_xproj, DI, DI + 2 * DS, n);
    n = NL * DI * DI;
    transpose_to_f16_kernel<<<(n + 255) / 256, 256, 0, stream>>>(dt_w, t_dt, DI, DI, n);
    n = NL * DI * DM;
    transpose_to_f16_kernel<<<(n + 255) / 256, 256, 0, stream>>>(out_w, t_out, DI, DM, n);
  }

  // Embedding + positional encoding
  {
    int n = ROWS * DM;
    embed_kernel<<<(n + 255) / 256, 256, 0, stream>>>(x, inp_w, inp_b, wh, wh16);
  }

  const int elemsDI = ROWS * DI;
  for (int lay = 0; lay < NL; ++lay) {
    // 1) xz = h @ in_proj_w          (8192 x 128 x 512)
    gemm_wmma_f16<DM, DM, 0, false><<<dim3((2 * DI) / 32, ROWS / 64), 128, 0, stream>>>(
        wh16, t_inproj + (size_t)lay * DM * 2 * DI,
        wxz, 2 * DI, nullptr, nullptr, 0);

    // 2) depthwise causal conv + SiLU
    conv_silu_kernel<<<(elemsDI + 255) / 256, 256, 0, stream>>>(
        wxz, conv_w + (size_t)lay * DI * DC, conv_b + (size_t)lay * DI, wxc, wxc16);

    // 3) proj = xc @ xproj_w         (8192 x 256 x 288), f16 shadow of first 256
    gemm_wmma_f16<DI, DI, 0, true><<<dim3((DI + 2 * DS) / 32, ROWS / 64), 128, 0, stream>>>(
        wxc16, t_xproj + (size_t)lay * DI * (DI + 2 * DS),
        wproj, DI + 2 * DS, nullptr, wpj16, DI);

    // 4) delta = softplus(proj[:, :256] @ dt_w + dt_b)   (8192 x 256 x 256)
    gemm_wmma_f16<DI, DI, 2, false><<<dim3(DI / 32, ROWS / 64), 128, 0, stream>>>(
        wpj16, t_dt + (size_t)lay * DI * DI,
        wdt, DI, dt_b + (size_t)lay * DI, nullptr, 0);

    // 5) chunked selective scan
    scan_pass1<<<dim3(NCHUNK, BB), DI, 0, stream>>>(
        wdt, wxc, wproj, A_log + (size_t)lay * DI * DS, wP, wQ);
    scan_pass2<<<BB, DI, 0, stream>>>(wP, wQ, wS0);
    scan_pass3<<<dim3(NCHUNK, BB), DI, 0, stream>>>(
        wdt, wxc, wproj, A_log + (size_t)lay * DI * DS,
        Dp + (size_t)lay * DI, wS0, wy);

    // 6) gate with silu(z) -> f16 operand for out-proj
    gate_kernel<<<(elemsDI + 255) / 256, 256, 0, stream>>>(wy, wxz, wyg16);

    // 7) out = yg @ out_w            (8192 x 256 x 128)
    gemm_wmma_f16<DI, DI, 0, false><<<dim3(DM / 32, ROWS / 64), 128, 0, stream>>>(
        wyg16, t_out + (size_t)lay * DI * DM,
        wout, DM, nullptr, nullptr, 0);

    // 8) h = LayerNorm(out + h)
    add_ln_kernel<<<ROWS, DM, 0, stream>>>(
        wout, wh, wh16, ln_w + (size_t)lay * DM, ln_b + (size_t)lay * DM);
  }

  // Pool + head
  mean_kernel<<<BB, DM, 0, stream>>>(wh, whm);
  head_kernel<<<BB, DM, 0, stream>>>(whm, cls_ln_w, cls_ln_b,
                                     cls_w1, cls_b1, cls_w2, cls_b2, out);
}
